// MSSA_26534307955021
// MI455X (gfx1250) — matrix operand order repaired and checked by reading, not compiled, not run on hardware
//
#include <hip/hip_runtime.h>
#include <hip/hip_bf16.h>

typedef __attribute__((ext_vector_type(16))) _Float16 v16h;
typedef __attribute__((ext_vector_type(8)))  _Float16 v8h;
typedef __attribute__((ext_vector_type(8)))  float    v8f;
typedef __attribute__((ext_vector_type(4)))  int      v4i;

constexpr int Bc = 2, Tc = 2048, Cc = 1024, Hc = 16, HDc = 64, RDc = 32;

#if defined(__has_builtin)
#if __has_builtin(__builtin_amdgcn_global_load_async_to_lds_b128)
#define USE_ASYNC_LDS 1
#else
#define USE_ASYNC_LDS 0
#endif
#else
#define USE_ASYNC_LDS 0
#endif

// ---------------------------------------------------------------------------
// WMMA fragment loaders (layouts per CDNA5 ISA §7.12.2, wave32).
// Every loader reads two contiguous 8-halfword runs per lane -> b128 loads.
// ---------------------------------------------------------------------------

// A fragment: 16x32 f16 tile from row-major [16][ld].
// lane: m = lane&15, half = lane>>4; VGPR v: k = 16*(v>=4) + 8*half + 2*(v&3)
__device__ inline v16h load_a_rm(const _Float16* p, int ld) {
  const int lane = threadIdx.x & 31;
  const int m = lane & 15, hf = lane >> 4;
  const _Float16* row = p + (size_t)m * ld;
  v16h a;
#pragma unroll
  for (int v = 0; v < 8; ++v) {
    const int k = ((v >> 2) << 4) + hf * 8 + ((v & 3) << 1);
    a[2 * v]     = row[k];
    a[2 * v + 1] = row[k + 1];
  }
  return a;
}

// B fragment: 32x16 (KxN) where storage is column-major: B[k][n] = p[n*ld + k].
// lane: n = lane&15, half = lane>>4; VGPR v: k = 16*(v>=4) + 8*half + 2*(v&3).
__device__ inline v16h load_b_cm(const _Float16* p, int ld) {
  const int lane = threadIdx.x & 31;
  const int n = lane & 15, hf = lane >> 4;
  const _Float16* row = p + (size_t)n * ld + hf * 8;
  v16h b;
#pragma unroll
  for (int v = 0; v < 8; ++v) {
    b[2 * v]     = row[2 * v + ((v >> 2) << 3)];
    b[2 * v + 1] = row[2 * v + 1 + ((v >> 2) << 3)];
  }
  return b;
}

__device__ inline v8f wmma_f16(v16h a, v16h b, v8f c) {
  return __builtin_amdgcn_wmma_f32_16x16x32_f16(false, a, false, b, (short)0, c,
                                                false, false);
}

// ---------------------------------------------------------------------------
// Kernel 1a: plain fp32 -> f16 cast (for x)
// ---------------------------------------------------------------------------
__global__ void cast_f32_f16(const float* __restrict__ src,
                             _Float16* __restrict__ dst, int n) {
  int i = blockIdx.x * blockDim.x + threadIdx.x;
  if (i < n) dst[i] = (_Float16)src[i];
}

// ---------------------------------------------------------------------------
// Kernel 1b: fused transpose + cast for weights.
// src: fp32 [R][Ccols] row-major -> dst: f16 [Ccols][R] row-major.
// block (32,8), grid (Ccols/32, R/32)
// ---------------------------------------------------------------------------
__global__ void __launch_bounds__(256)
transpose_cast(const float* __restrict__ src, _Float16* __restrict__ dst,
               int R, int Ccols) {
  __shared__ float tile[32][33];
  const int c0 = blockIdx.x * 32, r0 = blockIdx.y * 32;
  const int tx = threadIdx.x, ty = threadIdx.y;
#pragma unroll
  for (int i = 0; i < 32; i += 8)
    tile[ty + i][tx] = src[(size_t)(r0 + ty + i) * Ccols + c0 + tx];
  __syncthreads();
#pragma unroll
  for (int i = 0; i < 32; i += 8)
    dst[(size_t)(c0 + ty + i) * R + r0 + tx] = (_Float16)tile[tx][ty + i];
}

// ---------------------------------------------------------------------------
// Kernel 2/5: WMMA GEMM  C(M,N) fp32 = A(M,K) f16 @ B(K,N) f16.
// A row-major; B supplied TRANSPOSED (BT is N x K row-major).
// 256 threads = 8 waves, block tile M=128 x N=64.  All 8 waves consume the
// SAME 64-column B tile, so it is staged once per k-step in LDS (8x less L2
// traffic), double-buffered: global fetch of tile k+1 (async-to-LDS on the
// CDNA5 path, ASYNCcnt-tracked) overlaps the 4 WMMAs on tile k.
// LDS column stride 48 halfwords: keeps 16B alignment for b128 ops and
// avoids the 4-way bank conflict of a 64B stride.
// grid = (N/64, M/128); requires M%128==0, N%64==0, K%32==0.
// ---------------------------------------------------------------------------
__global__ void __launch_bounds__(256)
gemm_f16_wmma(const _Float16* __restrict__ A, const _Float16* __restrict__ BT,
              float* __restrict__ C, int M, int N, int K) {
  constexpr int BSTR = 48;                      // padded LDS stride (f16)
  __shared__ _Float16 bs[2][64 * BSTR];         // 12 KB

  const int tid = threadIdx.x;
  const int wave = tid >> 5, lane = tid & 31;
  const int m0 = (blockIdx.y * 8 + wave) * 16;
  const int n0 = blockIdx.x * 64;

  // cooperative B-tile fetch: thread tid moves 8 f16 (one b128)
  const int bn = tid >> 2;                      // tile column 0..63
  const int bk = (tid & 3) * 8;                 // k-chunk 0/8/16/24
  const _Float16* gsrc0 = BT + (size_t)(n0 + bn) * K + bk;

#if USE_ASYNC_LDS
  auto issueB = [&](int k0, int buf) {
    __builtin_amdgcn_global_load_async_to_lds_b128(
        (__attribute__((address_space(1))) v4i*)(gsrc0 + k0),
        (__attribute__((address_space(3))) v4i*)&bs[buf][bn * BSTR + bk],
        0, 0);
  };
  auto commitB = [&](int) {
    asm volatile("s_wait_asynccnt 0x0" ::: "memory");
  };
#else
  v8h breg;
  auto issueB = [&](int k0, int) { breg = *(const v8h*)(gsrc0 + k0); };
  auto commitB = [&](int buf) { *(v8h*)&bs[buf][bn * BSTR + bk] = breg; };
#endif

  issueB(0, 0);
  commitB(0);
  __syncthreads();

  v8f acc[4] = {{}, {}, {}, {}};
  v16h a = load_a_rm(A + (size_t)m0 * K, K);    // A prefetched one step ahead

  for (int k0 = 0; k0 < K; k0 += 32) {
    const int buf = (k0 >> 5) & 1;
    const bool more = (k0 + 32) < K;
    if (more) issueB(k0 + 32, buf ^ 1);

    v16h an = a;
    if (more) an = load_a_rm(A + (size_t)m0 * K + k0 + 32, K);

#pragma unroll
    for (int j = 0; j < 4; ++j) {
      v16h b = load_b_cm(&bs[buf][(j * 16) * BSTR], BSTR);
      acc[j] = wmma_f16(a, b, acc[j]);
    }
    a = an;

    if (more) commitB(buf ^ 1);
    __syncthreads();
  }

  const int hf = lane >> 4, nn = lane & 15;
#pragma unroll
  for (int j = 0; j < 4; ++j)
#pragma unroll
    for (int v = 0; v < 8; ++v) {
      const int m = v + 8 * hf;
      C[(size_t)(m0 + m) * N + n0 + j * 16 + nn] = acc[j][v];
    }
}

// ---------------------------------------------------------------------------
// Kernel 3: fused RoPE + RMS-norm.  One wave per (b,t,h) head of 64 values.
// Lane d (0..31) owns the rotary pair (d, d+32).
// Emits TWO f16 layouts: qn [B,H,T,HD] (Q A-frags + K^T B-frags) and
// qnT [B,H,HD,T] (V B-frags load contiguously along T).
// ---------------------------------------------------------------------------
__global__ void __launch_bounds__(256)
rope_rms(const float* __restrict__ w, const float* __restrict__ cosT,
         const float* __restrict__ sinT, _Float16* __restrict__ qn,
         _Float16* __restrict__ qnT) {
  const int head = blockIdx.x * 8 + (threadIdx.x >> 5);
  const int lane = threadIdx.x & 31;
  const int h = head % Hc;
  const int t = (head / Hc) % Tc;
  const int b = head / (Hc * Tc);

  const float* src = w + ((size_t)(b * Tc + t) * Hc + h) * HDc;
  const float x1 = src[lane];
  const float x2 = src[lane + RDc];
  const float c = cosT[t * RDc + lane];
  const float s = sinT[t * RDc + lane];
  const float y1 = x1 * c + x2 * s;
  const float y2 = -x1 * s + x2 * c;

  float ss = y1 * y1 + y2 * y2;
#pragma unroll
  for (int m = 16; m >= 1; m >>= 1) ss += __shfl_xor(ss, m, 32);
  const float r = rsqrtf(ss * (1.0f / 64.0f) + 1.1920929e-7f);

  const _Float16 z1 = (_Float16)(y1 * r);
  const _Float16 z2 = (_Float16)(y2 * r);

  _Float16* dst = qn + (((size_t)b * Hc + h) * Tc + t) * HDc;
  dst[lane]       = z1;
  dst[lane + RDc] = z2;

  _Float16* dstT = qnT + (((size_t)b * Hc + h) * HDc) * Tc + t;
  dstT[(size_t)lane * Tc]         = z1;
  dstT[(size_t)(lane + RDc) * Tc] = z2;
}

// ---------------------------------------------------------------------------
// Kernel 4: sliding-window flash attention.  One wave per 16-query tile of one
// (b,h).  32 keys per iteration: S = Q·K^T (4 WMMA), online softmax on the
// C-fragment (row = v + 8*half, col = lane&15 -> shfl_xor row-reductions),
// P routed through LDS (within-wave DS ordering) into A-fragment layout,
// then O += P·V (4 WMMA, V fragments from the d-major copy).
// T % 32 == 0 and j0 is 32-aligned with j0 <= q0+15 <= T-1, so key-row
// accesses never exceed T-1; out-of-window entries are masked to -1e30.
// ---------------------------------------------------------------------------
__global__ void __launch_bounds__(128)
attn_win(const _Float16* __restrict__ qn, const _Float16* __restrict__ qnT,
         _Float16* __restrict__ oh, const int* __restrict__ winp) {
  __shared__ _Float16 lds[4][16 * 32];

  const int wave = threadIdx.x >> 5;
  const int lane = threadIdx.x & 31;
  const int n = lane & 15, hf = lane >> 4;

  const int tile = blockIdx.x * 4 + wave;        // [0, B*H*(T/16))
  const int qt = tile % (Tc / 16);
  const int h  = (tile / (Tc / 16)) % Hc;
  const int b  = tile / ((Tc / 16) * Hc);
  const int q0 = qt * 16;

  const int window = *winp;
  const float scl = 0.125f;                       // HD^-0.5 = 1/8

  const _Float16* base  = qn  + (((size_t)b * Hc + h) * Tc) * HDc;  // [T][HD]
  const _Float16* baseT = qnT + (((size_t)b * Hc + h) * HDc) * Tc;  // [HD][T]

  const v16h qa0 = load_a_rm(base + (size_t)q0 * HDc, HDc);
  const v16h qa1 = load_a_rm(base + (size_t)q0 * HDc + 32, HDc);

  v8f oacc[4] = {{}, {}, {}, {}};
  float mrun[8], lrun[8];
#pragma unroll
  for (int v = 0; v < 8; ++v) { mrun[v] = -1e30f; lrun[v] = 0.0f; }

  int jstart = window > 0 ? q0 - window : 0;
  if (jstart < 0) jstart = 0;
  jstart &= ~31;

  for (int j0 = jstart; j0 <= q0 + 15; j0 += 32) {
    // --- S = scale * Q K^T for 32 keys (two 16-key fragments) ---
    v8f s0 = {}, s1 = {};
    {
      const _Float16* kp = base + (size_t)j0 * HDc;
      v16h b0 = load_b_cm(kp, HDc);
      v16h b1 = load_b_cm(kp + 32, HDc);
      s0 = wmma_f16(qa0, b0, s0);
      s0 = wmma_f16(qa1, b1, s0);
      v16h b2 = load_b_cm(kp + 16 * HDc, HDc);
      v16h b3 = load_b_cm(kp + 16 * HDc + 32, HDc);
      s1 = wmma_f16(qa0, b2, s1);
      s1 = wmma_f16(qa1, b3, s1);
    }

    // --- mask: allowed iff 0 <= i-j (and <= window when window>0) ---
#pragma unroll
    for (int v = 0; v < 8; ++v) {
      const int i = q0 + v + 8 * hf;
      const int d0 = i - (j0 + n);
      const int d1 = d0 - 16;
      const bool ok0 = (d0 >= 0) && (window <= 0 || d0 <= window);
      const bool ok1 = (d1 >= 0) && (window <= 0 || d1 <= window);
      s0[v] = ok0 ? s0[v] * scl : -1e30f;
      s1[v] = ok1 ? s1[v] * scl : -1e30f;
    }

    // --- online softmax: per-row max/sum via 16-lane half reductions ---
    float scale[8];
#pragma unroll
    for (int v = 0; v < 8; ++v) {
      float t = fmaxf(s0[v], s1[v]);
#pragma unroll
      for (int msk = 8; msk >= 1; msk >>= 1) t = fmaxf(t, __shfl_xor(t, msk, 32));
      const float mnew = fmaxf(mrun[v], t);
      scale[v] = __expf(mrun[v] - mnew);
      mrun[v] = mnew;
    }

#pragma unroll
    for (int v = 0; v < 8; ++v) {
      const float p0 = __expf(s0[v] - mrun[v]);
      const float p1 = __expf(s1[v] - mrun[v]);
      float t = p0 + p1;
#pragma unroll
      for (int msk = 8; msk >= 1; msk >>= 1) t += __shfl_xor(t, msk, 32);
      lrun[v] = lrun[v] * scale[v] + t;
      const int m = v + 8 * hf;
      lds[wave][m * 32 + n]      = (_Float16)p0;   // P in 16x32 row-major
      lds[wave][m * 32 + 16 + n] = (_Float16)p1;
    }

    // --- rescale O, then O += P·V ---
#pragma unroll
    for (int c = 0; c < 4; ++c)
#pragma unroll
      for (int v = 0; v < 8; ++v) oacc[c][v] *= scale[v];

    const v16h pA = load_a_rm(&lds[wave][0], 32);  // single-wave LDS RAW: DS in-order
#pragma unroll
    for (int c = 0; c < 4; ++c) {
      v16h vB = load_b_cm(baseT + (size_t)(c * 16) * Tc + j0, Tc);
      oacc[c] = wmma_f16(pA, vB, oacc[c]);
    }
  }

  // --- normalize and store f16 [B,T,H,HD] ---
  float inv[8];
#pragma unroll
  for (int v = 0; v < 8; ++v) inv[v] = 1.0f / lrun[v];
#pragma unroll
  for (int c = 0; c < 4; ++c)
#pragma unroll
    for (int v = 0; v < 8; ++v) {
      const int m = v + 8 * hf;
      oh[(((size_t)b * Tc + q0 + m) * Hc + h) * HDc + c * 16 + n] =
          (_Float16)(oacc[c][v] * inv[v]);
    }
}

// ---------------------------------------------------------------------------
// Host-side launch
// ---------------------------------------------------------------------------
extern "C" void kernel_launch(void* const* d_in, const int* in_sizes, int n_in,
                              void* d_out, int out_size, void* d_ws,
                              size_t ws_size, hipStream_t stream) {
  const float* x      = (const float*)d_in[0];
  const float* cosT   = (const float*)d_in[1];
  const float* sinT   = (const float*)d_in[2];
  const float* W_qkv  = (const float*)d_in[3];
  const float* W_proj = (const float*)d_in[4];
  const int*   winp   = (const int*)d_in[5];
  float* out = (float*)d_out;

  const size_t MT = (size_t)Bc * Tc;   // 4096 rows
  // workspace layout (f16 unless noted)
  _Float16* xh      = (_Float16*)d_ws;                     // x        8 MiB
  _Float16* wqkvT   = xh + MT * Cc;                        // W_qkv^T  2 MiB
  _Float16* wprojT  = wqkvT + (size_t)Cc * Cc;             // W_proj^T 2 MiB
  float*    w32     = (float*)(wprojT + (size_t)Cc * Cc);  // fp32 w  16 MiB
  _Float16* qnh     = (_Float16*)(w32 + MT * Cc);          // [B,H,T,HD] 8 MiB
  _Float16* qnTh    = qnh + MT * Cc;                       // [B,H,HD,T] 8 MiB
  _Float16* ohb     = qnTh + MT * Cc;                      // [B,T,H,HD] 8 MiB

  // 1) casts (+ weight transposes)
  {
    int n = (int)(MT * Cc);
    cast_f32_f16<<<(n + 255) / 256, 256, 0, stream>>>(x, xh, n);
    dim3 tb(32, 8), tg(Cc / 32, Cc / 32);
    transpose_cast<<<tg, tb, 0, stream>>>(W_qkv, wqkvT, Cc, Cc);
    transpose_cast<<<tg, tb, 0, stream>>>(W_proj, wprojT, Cc, Cc);
  }

  // 2) w = x @ W_qkv   (4096x1024x1024, fp32 out)
  gemm_f16_wmma<<<dim3(Cc / 64, (int)(MT / 128)), 256, 0, stream>>>(
      xh, wqkvT, w32, (int)MT, Cc, Cc);

  // 3) RoPE + RMS-norm -> f16 [B,H,T,HD] and [B,H,HD,T]
  rope_rms<<<(Bc * Tc * Hc) / 8, 256, 0, stream>>>(w32, cosT, sinT, qnh, qnTh);

  // 4) sliding-window flash attention -> f16 [B,T,H,HD]
  attn_win<<<(Bc * Hc * (Tc / 16)) / 4, 128, 0, stream>>>(qnh, qnTh, ohb, winp);

  // 5) out = o @ W_proj  (fp32 out)
  gemm_f16_wmma<<<dim3(Cc / 64, (int)(MT / 128)), 256, 0, stream>>>(
      ohb, wprojT, out, (int)MT, Cc, Cc);
}
